// VectorQuantizerEMA_19928648253524
// MI455X (gfx1250) — matrix operand order repaired and checked by reading, not compiled, not run on hardware
//
#include <hip/hip_runtime.h>
#include <math.h>

typedef __attribute__((ext_vector_type(16))) __bf16 v16bf;
typedef __attribute__((ext_vector_type(8)))  float  v8f;

#define NROWS   131072          // 16*8192
#define DIM     128
#define KCODES  1024
#define NTILES  (KCODES / 16)   // 64
#define DECAYF  0.99f
#define OMD     0.01f           // 1 - decay
#define EPSF    1e-5f

// ---- d_out layout (floats, concatenated in reference return order) ----
#define OFF_Q     0
#define OFF_LOSS  (NROWS * DIM)        // 16777216
#define OFF_PERP  (OFF_LOSS + 1)
#define OFF_IDX   (OFF_PERP + 1)
#define OFF_NCS   (OFF_IDX + NROWS)
#define OFF_NEMA  (OFF_NCS + KCODES)
#define OFF_UPD   (OFF_NEMA + KCODES * DIM)

// ---- ws layout (bytes) ----
#define WS_EBF    0                 // KCODES*DIM __bf16   = 262144
#define WS_ENORM  262144            // KCODES float        = 4096
#define WS_CNT    266240            // KCODES float        = 4096
#define WS_ESUM   270336            // KCODES*DIM float    = 524288
#define WS_SCAL   794624            // loss, n, entropy (+pad) = 16
#define WS_OFFS   794640            // 1025 int (+pad)     = 4112
#define WS_CURS   798752            // KCODES int          = 4096
#define WS_IDXI   802848            // NROWS int           = 524288
#define WS_ORD    1327136           // NROWS int           = 524288
                                    // total ~1.77 MB

// ---------------------------------------------------------------------------
__global__ void vq_zero_kernel(float* counts, float* scal) {
    int i = blockIdx.x * blockDim.x + threadIdx.x;
    if (i < KCODES) counts[i] = 0.0f;
    if (i < 3)      scal[i]   = 0.0f;
}

// One block (128 thr) per code row: bf16 convert + squared norm.
__global__ void vq_prep_kernel(const float* __restrict__ E,
                               __bf16* __restrict__ ebf,
                               float* __restrict__ embN) {
    int k = blockIdx.x, t = threadIdx.x;
    float e = E[k * DIM + t];
    ebf[k * DIM + t] = (__bf16)e;
    float p = e * e;
    #pragma unroll
    for (int off = 16; off; off >>= 1) p += __shfl_xor(p, off, 32);
    __shared__ float part[4];
    if ((t & 31) == 0) part[t >> 5] = p;
    __syncthreads();
    if (t == 0) embN[k] = part[0] + part[1] + part[2] + part[3];
}

// A operand builder: 16x32 bf16 chunk c of a row, contiguous K order
// (lanes 0-15: K [32c..32c+15]; lanes 16-31: K [32c+16..32c+31])
__device__ __forceinline__ v16bf vq_make_a(const float* __restrict__ xrow,
                                           int c, int hf) {
    const int s0 = 32 * c + hf * 16;
    float4 f0 = *(const float4*)(xrow + s0);
    float4 f1 = *(const float4*)(xrow + s0 + 4);
    float4 f2 = *(const float4*)(xrow + s0 + 8);
    float4 f3 = *(const float4*)(xrow + s0 + 12);
    v16bf a;
    a[0]=(__bf16)f0.x;  a[1]=(__bf16)f0.y;  a[2]=(__bf16)f0.z;  a[3]=(__bf16)f0.w;
    a[4]=(__bf16)f1.x;  a[5]=(__bf16)f1.y;  a[6]=(__bf16)f1.z;  a[7]=(__bf16)f1.w;
    a[8]=(__bf16)f2.x;  a[9]=(__bf16)f2.y;  a[10]=(__bf16)f2.z; a[11]=(__bf16)f2.w;
    a[12]=(__bf16)f3.x; a[13]=(__bf16)f3.y; a[14]=(__bf16)f3.z; a[15]=(__bf16)f3.w;
    return a;
}

#define WMMA_BF16(Acc, Aop, Bop) \
    __builtin_amdgcn_wmma_f32_16x16x32_bf16(false, (Aop), false, (Bop), \
                                            (short)0, (Acc), false, false)

// ---------------------------------------------------------------------------
// Main kernel: 256 threads = 8 waves; each wave owns 32 rows (two 16-row
// A-sets sharing every B tile -> 8 WMMAs per 4KB of B traffic).
__global__ void __launch_bounds__(256)
vq_main_kernel(const float* __restrict__ X,
               const __bf16* __restrict__ ebf,
               const float*  __restrict__ embN,
               const float*  __restrict__ E,      // fp32 embeddings for gather
               float* __restrict__ qOut,          // d_out + OFF_Q
               float* __restrict__ idxOutF,       // d_out + OFF_IDX (as float)
               int*   __restrict__ idxInt,        // ws
               float* __restrict__ counts,
               float* __restrict__ lossAcc) {
    const int wid  = threadIdx.x >> 5;
    const int lane = threadIdx.x & 31;
    const int col  = lane & 15;        // N-column within a 16-code tile
    const int hf   = lane >> 4;        // half-wave selector for K runs
    const int rowBase = blockIdx.x * 256 + wid * 32;

    const float* xr0 = X + (size_t)(rowBase + col) * DIM;
    const float* xr1 = X + (size_t)(rowBase + 16 + col) * DIM;
    v16bf A0[4], A1[4];
    #pragma unroll
    for (int c = 0; c < 4; ++c) {
        A0[c] = vq_make_a(xr0, c, hf);
        A1[c] = vq_make_a(xr1, c, hf);
    }

    float best0[8], best1[8];
    int   bidx0[8], bidx1[8];
    #pragma unroll
    for (int r = 0; r < 8; ++r) {
        best0[r] = 3.4e38f; best1[r] = 3.4e38f;
        bidx0[r] = 0;       bidx1[r] = 0;
    }

    #pragma unroll 2
    for (int t = 0; t < NTILES; ++t) {
        // B operand: lane holds 16 contiguous dims of code row (16t + col);
        // lanes 16-31 cover K=16..31 of each 32-chunk -> +hf*16.
        const __bf16* brow = ebf + (size_t)(t * 16 + col) * DIM + hf * 16;
        v16bf B0 = *(const v16bf*)(brow);
        v16bf B1 = *(const v16bf*)(brow + 32);
        v16bf B2 = *(const v16bf*)(brow + 64);
        v16bf B3 = *(const v16bf*)(brow + 96);

        v8f acc0 = {};
        v8f acc1 = {};
        acc0 = WMMA_BF16(acc0, A0[0], B0);
        acc1 = WMMA_BF16(acc1, A1[0], B0);
        acc0 = WMMA_BF16(acc0, A0[1], B1);
        acc1 = WMMA_BF16(acc1, A1[1], B1);
        acc0 = WMMA_BF16(acc0, A0[2], B2);
        acc1 = WMMA_BF16(acc1, A1[2], B2);
        acc0 = WMMA_BF16(acc0, A0[3], B3);
        acc1 = WMMA_BF16(acc1, A1[3], B3);

        const int   code = t * 16 + col;
        const float en   = embN[code];
        #pragma unroll
        for (int r = 0; r < 8; ++r) {
            float s0 = fmaf(-2.0f, acc0[r], en);   // ||x||^2 invariant to argmin
            float s1 = fmaf(-2.0f, acc1[r], en);
            if (s0 < best0[r]) { best0[r] = s0; bidx0[r] = code; }
            if (s1 < best1[r]) { best1[r] = s1; bidx1[r] = code; }
        }
    }

    // ---- argmin across the 16 lanes of each half (xor masks < 16) ----
    #pragma unroll
    for (int r = 0; r < 8; ++r) {
        #pragma unroll
        for (int m = 1; m <= 8; m <<= 1) {
            float os0 = __shfl_xor(best0[r], m, 32);
            int   oi0 = __shfl_xor(bidx0[r], m, 32);
            if (os0 < best0[r]) { best0[r] = os0; bidx0[r] = oi0; }
            float os1 = __shfl_xor(best1[r], m, 32);
            int   oi1 = __shfl_xor(bidx1[r], m, 32);
            if (os1 < best1[r]) { best1[r] = os1; bidx1[r] = oi1; }
        }
    }

    // ---- gather / outputs (wave-cooperative, 4 dims per lane) ----
    float lossLocal = 0.0f;
    #pragma unroll 1
    for (int m = 0; m < 32; ++m) {
        const int idx = (m < 16)
            ? __shfl(bidx0[m & 7], (m >> 3) * 16, 32)
            : __shfl(bidx1[m & 7], (((m - 16) >> 3) & 1) * 16, 32);
        const int row = rowBase + m;
        const float4 e = ((const float4*)(E + (size_t)idx * DIM))[lane];
        const float4 x = ((const float4*)(X + (size_t)row * DIM))[lane];
        ((float4*)(qOut + (size_t)row * DIM))[lane] = e;  // straight-through value
        float dx = e.x - x.x, dy = e.y - x.y, dz = e.z - x.z, dw = e.w - x.w;
        lossLocal += dx * dx + dy * dy + dz * dz + dw * dw;
        if (lane == 0) {
            idxOutF[row] = (float)idx;
            idxInt[row]  = idx;
            atomicAdd(&counts[idx], 1.0f);
        }
    }
    #pragma unroll
    for (int off = 16; off; off >>= 1) lossLocal += __shfl_xor(lossLocal, off, 32);
    if (lane == 0) atomicAdd(lossAcc, lossLocal);
}

// ---------------------------------------------------------------------------
// Single-block 1024-wide exclusive prefix sum of counts -> offsets + cursors.
__global__ void vq_prefix_kernel(const float* __restrict__ counts,
                                 int* __restrict__ offs,
                                 int* __restrict__ cursor) {
    __shared__ int s[KCODES];
    int t = threadIdx.x;
    int c = (int)counts[t];
    s[t] = c;
    __syncthreads();
    #pragma unroll
    for (int d = 1; d < KCODES; d <<= 1) {
        int v = (t >= d) ? s[t - d] : 0;
        __syncthreads();
        s[t] += v;
        __syncthreads();
    }
    int incl = s[t];
    int excl = incl - c;
    offs[t]   = excl;
    cursor[t] = excl;
    if (t == KCODES - 1) offs[KCODES] = incl;
}

// Counting-sort scatter: row ids grouped by code.
__global__ void vq_scatter_kernel(const int* __restrict__ idxInt,
                                  int* __restrict__ cursor,
                                  int* __restrict__ order) {
    int i = blockIdx.x * blockDim.x + threadIdx.x;
    if (i >= NROWS) return;
    int k = idxInt[i];
    int pos = atomicAdd(&cursor[k], 1);
    order[pos] = i;
}

// Atomic-free embed_sum: one block (128 thr = one dim each) per code.
__global__ void vq_esum_kernel(const float* __restrict__ X,
                               const int* __restrict__ order,
                               const int* __restrict__ offs,
                               float* __restrict__ esum) {
    int k = blockIdx.x, t = threadIdx.x;
    float acc = 0.0f;
    int b = offs[k], e = offs[k + 1];
    for (int j = b; j < e; ++j) {
        int row = order[j];
        acc += X[(size_t)row * DIM + t];   // coalesced across the block
    }
    esum[k * DIM + t] = acc;
}

// ---------------------------------------------------------------------------
__global__ void vq_cluster_kernel(const float* __restrict__ CS,
                                  const float* __restrict__ counts,
                                  float* __restrict__ ncsOut,
                                  float* __restrict__ nAcc,
                                  float* __restrict__ entAcc) {
    int k = blockIdx.x * blockDim.x + threadIdx.x;
    if (k >= KCODES) return;
    float cnt = counts[k];
    float ncs = CS[k] * DECAYF + OMD * cnt;
    ncsOut[k] = ncs;
    atomicAdd(nAcc, ncs);
    float p = cnt * (1.0f / (float)NROWS);
    atomicAdd(entAcc, p * logf(p + 1e-10f));
}

__global__ void vq_scalar_kernel(const float* lossAcc, const float* entAcc,
                                 float* lossOut, float* perpOut) {
    lossOut[0] = 0.25f * lossAcc[0] * (1.0f / 16777216.0f);  // /(N*D), *commitment
    perpOut[0] = expf(-entAcc[0]);
}

__global__ void vq_ema_update_kernel(const float* __restrict__ EW,
                                     const float* __restrict__ esum,
                                     const float* __restrict__ ncs,
                                     const float* __restrict__ nAcc,
                                     float* __restrict__ nemaOut,
                                     float* __restrict__ updOut) {
    int i = blockIdx.x * blockDim.x + threadIdx.x;
    if (i >= KCODES * DIM) return;
    int k = i >> 7;
    float nem = EW[i] * DECAYF + OMD * esum[i];
    nemaOut[i] = nem;
    float n = nAcc[0];
    float smoothed = (ncs[k] + EPSF) / (n + (float)KCODES * EPSF) * n;
    updOut[i] = nem / smoothed;
}

// ---------------------------------------------------------------------------
extern "C" void kernel_launch(void* const* d_in, const int* in_sizes, int n_in,
                              void* d_out, int out_size, void* d_ws, size_t ws_size,
                              hipStream_t stream) {
    const float* X  = (const float*)d_in[0];   // [16,8192,128]
    const float* E  = (const float*)d_in[1];   // [1024,128]
    const float* CS = (const float*)d_in[2];   // [1024]
    const float* EW = (const float*)d_in[3];   // [1024,128]
    float* out = (float*)d_out;
    char*  ws  = (char*)d_ws;

    __bf16* ebf     = (__bf16*)(ws + WS_EBF);
    float*  embN    = (float*)(ws + WS_ENORM);
    float*  counts  = (float*)(ws + WS_CNT);
    float*  esum    = (float*)(ws + WS_ESUM);
    float*  lossAcc = (float*)(ws + WS_SCAL);
    float*  nAcc    = lossAcc + 1;
    float*  entAcc  = lossAcc + 2;
    int*    offs    = (int*)(ws + WS_OFFS);
    int*    cursor  = (int*)(ws + WS_CURS);
    int*    idxInt  = (int*)(ws + WS_IDXI);
    int*    order   = (int*)(ws + WS_ORD);

    vq_zero_kernel<<<4, 256, 0, stream>>>(counts, lossAcc);
    vq_prep_kernel<<<KCODES, DIM, 0, stream>>>(E, ebf, embN);
    vq_main_kernel<<<NROWS / 256, 256, 0, stream>>>(X, ebf, embN, E,
                                                    out + OFF_Q, out + OFF_IDX,
                                                    idxInt, counts, lossAcc);
    vq_prefix_kernel<<<1, KCODES, 0, stream>>>(counts, offs, cursor);
    vq_scatter_kernel<<<NROWS / 256, 256, 0, stream>>>(idxInt, cursor, order);
    vq_esum_kernel<<<KCODES, DIM, 0, stream>>>(X, order, offs, esum);
    vq_cluster_kernel<<<(KCODES + 255) / 256, 256, 0, stream>>>(CS, counts,
                                                                out + OFF_NCS, nAcc, entAcc);
    vq_scalar_kernel<<<1, 1, 0, stream>>>(lossAcc, entAcc, out + OFF_LOSS, out + OFF_PERP);
    vq_ema_update_kernel<<<(KCODES * DIM + 255) / 256, 256, 0, stream>>>(
        EW, esum, out + OFF_NCS, nAcc, out + OFF_NEMA, out + OFF_UPD);
}
